// ReadoutLayer_50543175139486
// MI455X (gfx1250) — compile-verified, matrix operand order checked by reading, not run on hardware
//
#include <hip/hip_runtime.h>

typedef __attribute__((ext_vector_type(16))) _Float16 v16h;
typedef __attribute__((ext_vector_type(8)))  float    v8f;

#define NH 64
#define KF 96      /* padded K for layer-1 (66 -> 96, 3 chunks of 32) */
#define RN 128
#define TM 16      /* node-tile rows (WMMA M) */

/* ---- packed f16 weight buffer (global, produced by pack_weights) ---- */
#define W_I1T    0                    /* [128][96]  transposed, K padded */
#define W_J1T    (W_I1T + RN*KF)      /* 12288 */
#define W_I2T    (W_J1T + RN*KF)      /* 24576 */
#define W_J2T    (W_I2T + RN*RN)      /* 40960 */
#define W_TOTAL  (W_J2T + RN*RN)      /* 57344 halfs = 112 KB */

/* ---- LDS layout (half-element offsets): activations only ---- */
#define OFF_HX   0
#define OFF_H1I  (OFF_HX  + TM*KF)    /* 1536 */
#define OFF_H1J  (OFF_H1I + TM*RN)    /* 3584 */
#define SMEM_HALFS   (OFF_H1J + TM*RN)    /* 5632 halfs = 11264 B */
#define SMEM_F32_OFF (SMEM_HALFS*2)
#define SMEM_BYTES   (SMEM_F32_OFF + (RN+RN)*4)   /* + pooled[128], red[128] */

__device__ __forceinline__ float fast_rcp(float x) { return __builtin_amdgcn_rcpf(x); }

__device__ __forceinline__ float tanh_fast(float x) {
  /* branch-free: 1 - 2/(exp(2x)+1); saturates correctly at +-1 */
  float e = __expf(2.f * x);
  return 1.f - 2.f * fast_rcp(e + 1.f);
}

__device__ __forceinline__ float sigmoid_fast(float x) {
  return fast_rcp(1.f + __expf(-x));
}

__device__ __forceinline__ float selu_f(float x) {
  const float a = 1.6732632423543772f, s = 1.0507009873554805f;
  return s * (x > 0.f ? x : a * (__expf(x) - 1.f));
}

/* Load one 16x32 f16 WMMA fragment (works for LDS or global pointers).
   `base` already includes the per-lane row/col offset (row-major, K contiguous).
   ISA layout: lanes 0-15 -> K {0..7,16..23}+kc32, lanes 16-31 -> K {8..15,24..31}+kc32,
   VGPR v holds the (k,k+1) half-pair -> one dword load each. */
__device__ __forceinline__ v16h frag_ld(const _Float16* base, int kc32, int hi) {
  union { unsigned u[8]; v16h h; } f;
  const unsigned* p = (const unsigned*)base;
#pragma unroll
  for (int v = 0; v < 8; ++v) {
    int k = kc32 + ((v & 4) << 2) + (hi << 3) + ((v & 3) << 1);
    f.u[v] = p[k >> 1];
  }
  return f.h;
}

#define WMMA(a, b, c) \
  __builtin_amdgcn_wmma_f32_16x16x32_f16(false, (a), false, (b), (short)0, (c), false, false)

/* ---------- prep: f32 weights -> f16, transposed [N][K], K padded ---------- */
__global__ void pack_weights(const float* __restrict__ i1w, const float* __restrict__ j1w,
                             const float* __restrict__ i2w, const float* __restrict__ j2w,
                             _Float16* __restrict__ gw) {
  int idx = blockIdx.x * blockDim.x + threadIdx.x;
  if (idx >= W_TOTAL) return;
  float v = 0.f;
  if (idx < RN * KF) {                     /* i1t[n][k] = i1w[k][n] */
    int n = idx / KF, k = idx - n * KF;
    if (k < NH + 2) v = i1w[k * RN + n];
  } else if (idx < 2 * RN * KF) {          /* j1t */
    int l = idx - RN * KF;
    int n = l / KF, k = l - n * KF;
    if (k < NH + 2) v = j1w[k * RN + n];
  } else if (idx < 2 * RN * KF + RN * RN) { /* i2t */
    int l = idx - 2 * RN * KF;
    int n = l / RN, k = l - n * RN;
    v = i2w[k * RN + n];
  } else {                                  /* j2t */
    int l = idx - 2 * RN * KF - RN * RN;
    int n = l / RN, k = l - n * RN;
    v = j2w[k * RN + n];
  }
  gw[idx] = (_Float16)v;
}

/* ---------- main: one workgroup (8 waves) per graph ---------- */
__global__ __launch_bounds__(256)
void readout_main(const float* __restrict__ nodes,
                  const float* __restrict__ node_mask,
                  const int*   __restrict__ n_node,
                  const float* __restrict__ i1b, const float* __restrict__ i2b,
                  const float* __restrict__ j1b, const float* __restrict__ j2b,
                  const float* __restrict__ h1w, const float* __restrict__ h1b,
                  const float* __restrict__ h2w, const float* __restrict__ h2b,
                  const _Float16* __restrict__ gw,
                  float* __restrict__ out, int Ntot, int numG) {
  extern __shared__ char smem[];
  _Float16* sh  = (_Float16*)smem;
  _Float16* hx  = sh + OFF_HX;
  _Float16* h1i = sh + OFF_H1I;
  _Float16* h1j = sh + OFF_H1J;
  float* pooled = (float*)(smem + SMEM_F32_OFF);
  float* red    = pooled + RN;

  const int tid  = threadIdx.x;
  const int g    = blockIdx.x;
  const int lane = tid & 31;
  const int wv   = tid >> 5;

  const int nper = n_node[0];
  int start = g * nper;
  int end   = (g == numG - 1) ? Ntot : start + nper;   /* jnp.repeat pad rule */
  if (end > Ntot) end = Ntot;

  const int colL = lane & 15;
  const int hi   = lane >> 4;
  const int nb   = wv * 16;                 /* this wave's output-column base */
  const float b_i1 = i1b[nb + colL];
  const float b_j1 = j1b[nb + colL];
  const float b_i2 = i2b[nb + colL];
  const float b_j2 = j2b[nb + colL];

  /* ---- hoist all loop-invariant B-matrix weight fragments into registers
     (14 frags x 8 VGPRs = 112 VGPRs), loaded once from L2-resident pack ---- */
  v16h Bi1[KF / 32], Bj1[KF / 32], Bi2[RN / 32], Bj2[RN / 32];
  {
    const _Float16* wb1i = gw + W_I1T + (nb + colL) * KF;
    const _Float16* wb1j = gw + W_J1T + (nb + colL) * KF;
    const _Float16* wb2i = gw + W_I2T + (nb + colL) * RN;
    const _Float16* wb2j = gw + W_J2T + (nb + colL) * RN;
#pragma unroll
    for (int kc = 0; kc < KF / 32; ++kc) {
      Bi1[kc] = frag_ld(wb1i, kc * 32, hi);
      Bj1[kc] = frag_ld(wb1j, kc * 32, hi);
    }
#pragma unroll
    for (int kc = 0; kc < RN / 32; ++kc) {
      Bi2[kc] = frag_ld(wb2i, kc * 32, hi);
      Bj2[kc] = frag_ld(wb2j, kc * 32, hi);
    }
  }

  float acc[8];
#pragma unroll
  for (int r = 0; r < 8; ++r) acc[r] = 0.f;

  for (int base = start; base < end; base += TM) {
    /* stage hx tile [16][96]: nodes[.,0:64] ++ nodes[.,0:2] ++ zero-pad, f32->f16 */
#pragma unroll
    for (int it = 0; it < (TM * KF) / 256; ++it) {
      int idx = it * 256 + tid;
      int row = idx / KF, col = idx - row * KF;
      int n = base + row;
      float v = 0.f;
      if (n < end) {
        if (col < NH)          v = nodes[(size_t)n * NH + col];
        else if (col < NH + 2) v = nodes[(size_t)n * NH + (col - NH)];
      }
      hx[idx] = (_Float16)v;
    }
    if (tid < TM) { /* prefetch next tile's node rows */
      int nn = base + TM + tid;
      if (nn < end) __builtin_prefetch(nodes + (size_t)nn * NH, 0, 0);
    }
    __syncthreads();

    /* GEMM1: [16x96] @ {i1,j1} -> activations -> f16 LDS (A frag shared by paths) */
    {
      const _Float16* aBase = hx + colL * KF;
      v8f ci = {}; v8f cj = {};
#pragma unroll
      for (int kc = 0; kc < KF / 32; ++kc) {
        v16h a = frag_ld(aBase, kc * 32, hi);
        ci = WMMA(a, Bi1[kc], ci);
        cj = WMMA(a, Bj1[kc], cj);
      }
#pragma unroll
      for (int r = 0; r < 8; ++r) {
        int row = r + hi * 8;                       /* C layout: M = r + 8*hi */
        float xi = tanh_fast(ci[r] + b_i1);
        float xj = selu_f(cj[r] + b_j1);
        h1i[row * RN + nb + colL] = (_Float16)xi;
        h1j[row * RN + nb + colL] = (_Float16)xj;
      }
    }
    __syncthreads();

    /* GEMM2: [16x128] @ {i2,j2} -> gate, mask, accumulate in C regs */
    {
      const _Float16* ai = h1i + colL * RN;
      const _Float16* aj = h1j + colL * RN;
      v8f di = {}; v8f dj = {};
#pragma unroll
      for (int kc = 0; kc < RN / 32; ++kc) {
        v16h a1 = frag_ld(ai, kc * 32, hi);
        di = WMMA(a1, Bi2[kc], di);
        v16h a2 = frag_ld(aj, kc * 32, hi);
        dj = WMMA(a2, Bj2[kc], dj);
      }
#pragma unroll
      for (int r = 0; r < 8; ++r) {
        int n = base + r + hi * 8;
        float m  = (n < end) ? node_mask[n] : 0.f;
        float si = sigmoid_fast(di[r] + b_i2);
        float jo = dj[r] + b_j2;
        acc[r] = __builtin_fmaf(si * jo, m, acc[r]);   /* segment-sum */
      }
    }
    __syncthreads();
  }

  /* reduce the 16 M-rows of the accumulated tile -> pooled[128] */
  float local = 0.f;
#pragma unroll
  for (int r = 0; r < 8; ++r) local += acc[r];
  local += __shfl_down(local, 16, 32);
  if (lane < 16) pooled[nb + lane] = local;
  __syncthreads();

  /* per-graph MLP: selu(pooled @ h1w + h1b) . h2w + h2b  (tiny, VALU) */
  if (tid < RN) {
    float a = h1b[tid];
#pragma unroll 8
    for (int k = 0; k < RN; ++k) a = __builtin_fmaf(pooled[k], h1w[k * RN + tid], a);
    red[tid] = selu_f(a) * h2w[tid];
  }
  __syncthreads();
  for (int s = 64; s > 0; s >>= 1) {
    if (tid < s) red[tid] += red[tid + s];
    __syncthreads();
  }
  if (tid == 0) out[g] = red[0] + h2b[0];
}

extern "C" void kernel_launch(void* const* d_in, const int* in_sizes, int n_in,
                              void* d_out, int out_size, void* d_ws, size_t ws_size,
                              hipStream_t stream) {
  const float* nodes     = (const float*)d_in[0];
  const float* node_mask = (const float*)d_in[1];
  const int*   n_node    = (const int*)  d_in[2];
  const float* i1w = (const float*)d_in[3];
  const float* i1b = (const float*)d_in[4];
  const float* i2w = (const float*)d_in[5];
  const float* i2b = (const float*)d_in[6];
  const float* j1w = (const float*)d_in[7];
  const float* j1b = (const float*)d_in[8];
  const float* j2w = (const float*)d_in[9];
  const float* j2b = (const float*)d_in[10];
  const float* h1w = (const float*)d_in[11];
  const float* h1b = (const float*)d_in[12];
  const float* h2w = (const float*)d_in[13];
  const float* h2b = (const float*)d_in[14];

  const int Ntot = in_sizes[0] / NH;
  const int numG = in_sizes[2];
  _Float16* gw = (_Float16*)d_ws;          /* 112 KB of scratch */

  pack_weights<<<(W_TOTAL + 255) / 256, 256, 0, stream>>>(i1w, j1w, i2w, j2w, gw);
  readout_main<<<numG, 256, SMEM_BYTES, stream>>>(
      nodes, node_mask, n_node, i1b, i2b, j1b, j2b,
      h1w, h1b, h2w, h2b, gw, (float*)d_out, Ntot, numG);
}
